// BioLSTM_58179626992363
// MI455X (gfx1250) — matrix operand order repaired and checked by reading, not compiled, not run on hardware
//
#include <hip/hip_runtime.h>
#include <hip/hip_bf16.h>

// ---------------------------------------------------------------------------
// BioLSTM on MI455X (gfx1250): bf16 WMMA path, fused per-step kernel
//   B=64, S=512, I=512, H=1024, O=256, TAU=0.1
// ---------------------------------------------------------------------------

typedef __attribute__((ext_vector_type(16))) __bf16 v16bf;
typedef __attribute__((ext_vector_type(8)))  float  v8f;

union Frag32 { uint4 u[2]; v16bf v; };   // 32 bytes = 16 bf16 per lane

static __device__ __forceinline__ float sigf(float x) {
  return 1.0f / (1.0f + __expf(-x));
}

// ---------------------------------------------------------------------------
// Pack a row-major [N x K] fp32 weight matrix into bf16 WMMA B-fragment
// order: dst[((kb*Ntiles + tn)*32 + lane)*16 + e]  where
//   k = kb*32 + (lane>>4)*8 + (e&7) + (e>>3)*16   (B 32x16 bf16 layout: table K order)
//   n = tn*16 + (lane&15)
// ---------------------------------------------------------------------------
__global__ void pack_b_kernel(const float* __restrict__ src,
                              __bf16* __restrict__ dst,
                              int K, int Ntiles, int total) {
  int t = blockIdx.x * blockDim.x + threadIdx.x;
  if (t >= total) return;
  int e    = t & 15;
  int lane = (t >> 4) & 31;
  int tn   = (t >> 9) % Ntiles;
  int kb   = t / (Ntiles << 9);
  int k = kb * 32 + ((lane >> 4) << 3) + (e & 7) + ((e >> 3) << 4);
  int n = tn * 16 + (lane & 15);
  dst[t] = (__bf16)src[(size_t)n * K + k];
}

// ---------------------------------------------------------------------------
// Combine gate biases (bWi + bWh + bgate -> bsum[4096]) and zero the
// recurrent state (h0 fp32, h0 bf16, c). 65536 threads.
// ---------------------------------------------------------------------------
__global__ void bias_init_kernel(const float* __restrict__ bWi,
                                 const float* __restrict__ bWh,
                                 const float* __restrict__ bgate,
                                 float* __restrict__ bsum,
                                 float* __restrict__ h0,
                                 __bf16* __restrict__ hb0,
                                 float* __restrict__ c) {
  int t = blockIdx.x * blockDim.x + threadIdx.x;
  if (t < 4096) bsum[t] = bWi[t] + bWh[t] + bgate[t];
  h0[t]  = 0.0f;
  c[t]   = 0.0f;
  hb0[t] = (__bf16)0.0f;
}

// ---------------------------------------------------------------------------
// Fully fused LSTM step (one launch per time-step):
//   gates[64,4096] = [x(:,s,:) | h] @ [WiT ; WhT]   (K = 512+1024, bf16 WMMA)
//   followed by the in-block LSTM pointwise update.
//
// Grid: 128 blocks x 256 thr.  block = (tile_m = bIdx>>5, p = bIdx&31).
// Wave w (0..7) computes n-tile  g*64 + jt,  g = w&3, jt = 2p + (w>>2),
// so the block owns ALL FOUR gates for rows [16*tile_m,+16) x j in [32p,+32).
// A (16 rows x 1536 bf16 = 48 KB) staged in LDS; gate tiles staged in LDS;
// h is double-buffered across steps (read hb_prev/hf_prev, write *_next).
// ---------------------------------------------------------------------------
__global__ __launch_bounds__(256) void
step_fused_kernel(const float* __restrict__ x,          // [B,S,I] fp32
                  const __bf16* __restrict__ hb_prev,   // [64,1024] bf16
                  const __bf16* __restrict__ Bp0,       // packed WiT (16 kb)
                  const __bf16* __restrict__ Bp1,       // packed WhT (32 kb)
                  const float* __restrict__ bsum,       // [4096]
                  const float* __restrict__ noises,     // [B,S,H] fp32
                  const float* __restrict__ b_h,        // [H]
                  float* __restrict__ c,                // [64,1024] in-place
                  const float* __restrict__ hf_prev,    // [64,1024]
                  float* __restrict__ hf_next,          // [64,1024]
                  __bf16* __restrict__ hb_next,         // [64,1024]
                  float* __restrict__ hid_out,          // [B,S,H]
                  int s) {
  __shared__ __bf16 ldsA[16 * 1536];   // 48 KB
  __shared__ float  ldsG[8 * 256];     // 8 KB: per-wave 16x16 gate tile

  const int S = 512, I = 512;
  int tile_m = blockIdx.x >> 5;        // 0..3
  int p      = blockIdx.x & 31;        // 0..31  (j block of 32)
  int w      = threadIdx.x >> 5;       // wave 0..7
  int lane   = threadIdx.x & 31;
  int g      = w & 3;                  // gate
  int jt     = 2 * p + (w >> 2);       // j tile (0..63)
  int tile_n = g * 64 + jt;            // 0..255

  // ---- stage A: x rows (fp32 -> bf16) then h rows (bf16 copy) -------------
  for (int idx = threadIdx.x; idx < 16 * 512; idx += 256) {
    int m = idx >> 9, k = idx & 511;
    int b = tile_m * 16 + m;
    ldsA[m * 1536 + k] = (__bf16)x[((size_t)b * S + s) * I + k];
  }
  for (int idx = threadIdx.x; idx < 16 * 1024; idx += 256) {
    int m = idx >> 10, k = idx & 1023;
    ldsA[m * 1536 + 512 + k] = hb_prev[(size_t)(tile_m * 16 + m) * 1024 + k];
  }
  __syncthreads();

  // A fragment addressing (16-bit A 16x32 layout)
  int mA  = lane & 15;
  int bkA = (lane >> 4) * 8;
  int abase = mA * 1536 + bkA;

  v8f acc = {};
  // ---- K part 1: x @ WiT (16 k-blocks over Bp0) ---------------------------
  {
    const __bf16* bptr = Bp0 + (((size_t)tile_n) * 32 + lane) * 16;
    for (int kb = 0; kb < 16; ++kb) {
      Frag32 afrag, bfrag;
      const uint4* ap = (const uint4*)&ldsA[abase + kb * 32];
      afrag.u[0] = ap[0];
      afrag.u[1] = ap[2];
      const uint4* bp = (const uint4*)bptr;
      bfrag.u[0] = bp[0];
      bfrag.u[1] = bp[1];
      if (kb < 15) __builtin_prefetch(bptr + 256 * 512, 0, 1);
      bptr += (size_t)256 * 512;
      acc = __builtin_amdgcn_wmma_f32_16x16x32_bf16(
          false, afrag.v, false, bfrag.v, (short)0, acc, false, false);
    }
  }
  // ---- K part 2: h @ WhT (32 k-blocks over Bp1) ---------------------------
  {
    const __bf16* bptr = Bp1 + (((size_t)tile_n) * 32 + lane) * 16;
    for (int kb = 16; kb < 48; ++kb) {
      Frag32 afrag, bfrag;
      const uint4* ap = (const uint4*)&ldsA[abase + kb * 32];
      afrag.u[0] = ap[0];
      afrag.u[1] = ap[2];
      const uint4* bp = (const uint4*)bptr;
      bfrag.u[0] = bp[0];
      bfrag.u[1] = bp[1];
      if (kb < 47) __builtin_prefetch(bptr + 256 * 512, 0, 1);
      bptr += (size_t)256 * 512;
      acc = __builtin_amdgcn_wmma_f32_16x16x32_bf16(
          false, afrag.v, false, bfrag.v, (short)0, acc, false, false);
    }
  }

  // ---- deposit gate tile (+ bias) into LDS --------------------------------
  // D layout: vgpr j -> row m = j + 8*(lane>=16), col n = lane&15
  int half = lane >> 4;
  int nloc = lane & 15;
  float bias = bsum[tile_n * 16 + nloc];
#pragma unroll
  for (int j = 0; j < 8; ++j) {
    ldsG[w * 256 + (j + half * 8) * 16 + nloc] = acc[j] + bias;
  }
  __syncthreads();

  // ---- in-block LSTM pointwise update -------------------------------------
  // block owns rows b in [16*tile_m,+16), j in [32p,+32): 512 elements.
  for (int idx = threadIdx.x; idx < 512; idx += 256) {
    int m   = idx >> 5;        // 0..15
    int jj  = idx & 31;        // 0..31
    int jtl = jj >> 4;         // which of the 2 j-tiles
    int jn  = jj & 15;
    int gofs = jtl * 4 * 256 + m * 16 + jn;
    float gi = ldsG[gofs + 0 * 256];
    float gf = ldsG[gofs + 1 * 256];
    float gg = ldsG[gofs + 2 * 256];
    float go = ldsG[gofs + 3 * 256];
    int b = tile_m * 16 + m;
    int j = (2 * p + jtl) * 16 + jn;
    size_t st = (size_t)b * 1024 + j;

    float i_t = sigf(gi);
    float f_t = sigf(gf);
    float g_t = tanhf(gg);
    float o_t = sigf(go);
    float cn = f_t * c[st] + i_t * g_t;
    float nh = o_t * tanhf(cn);
    float hv = hf_prev[st];
    float hn = hv + 0.1f * (-hv + nh + noises[((size_t)b * 512 + s) * 1024 + j] + b_h[j]);
    c[st]       = cn;
    hf_next[st] = hn;
    hb_next[st] = (__bf16)hn;
    hid_out[((size_t)b * 512 + s) * 1024 + j] = hn;
  }
}

// ---------------------------------------------------------------------------
// Output GEMM: outputs[32768,256] = hidden[32768,1024] @ W_outT + b_out
// Grid: 2048 blocks x 256 thr; block = 1 m-tile; each wave owns 2 n-tiles
// (two independent accumulator chains). A staged fp32->bf16 in LDS (32 KB).
// ---------------------------------------------------------------------------
__global__ __launch_bounds__(256) void
out_gemm_kernel(const float* __restrict__ hid,      // [B*S, H] fp32
                const __bf16* __restrict__ Bp2,     // packed W_outT (32 kb, 16 tn)
                const float* __restrict__ b_out,    // [O]
                float* __restrict__ out) {          // [B*S, O]
  __shared__ __bf16 ldsA[16 * 1024];
  int tile_m = blockIdx.x;                 // 0..2047
  int w      = threadIdx.x >> 5;
  int lane   = threadIdx.x & 31;
  int tn0    = w * 2;                      // this wave's two n-tiles

  for (int idx = threadIdx.x; idx < 16 * 1024; idx += 256) {
    int m = idx >> 10, k = idx & 1023;
    ldsA[idx] = (__bf16)hid[((size_t)tile_m * 16 + m) * 1024 + k];
  }
  __syncthreads();

  int mA  = lane & 15;
  int bkA = (lane >> 4) * 8;
  int abase = mA * 1024 + bkA;

  v8f acc0 = {}, acc1 = {};
  const __bf16* bptr0 = Bp2 + (((size_t)tn0 + 0) * 32 + lane) * 16;
  const __bf16* bptr1 = Bp2 + (((size_t)tn0 + 1) * 32 + lane) * 16;
  for (int kb = 0; kb < 32; ++kb) {
    Frag32 afrag, bf0, bf1;
    const uint4* ap = (const uint4*)&ldsA[abase + kb * 32];
    afrag.u[0] = ap[0];
    afrag.u[1] = ap[2];
    const uint4* bp0 = (const uint4*)bptr0;
    const uint4* bp1 = (const uint4*)bptr1;
    bf0.u[0] = bp0[0]; bf0.u[1] = bp0[1];
    bf1.u[0] = bp1[0]; bf1.u[1] = bp1[1];
    bptr0 += (size_t)16 * 512;
    bptr1 += (size_t)16 * 512;
    acc0 = __builtin_amdgcn_wmma_f32_16x16x32_bf16(
        false, afrag.v, false, bf0.v, (short)0, acc0, false, false);
    acc1 = __builtin_amdgcn_wmma_f32_16x16x32_bf16(
        false, afrag.v, false, bf1.v, (short)0, acc1, false, false);
  }

  int half = lane >> 4;
  int nloc = lane & 15;
#pragma unroll
  for (int j = 0; j < 8; ++j) {
    int r = tile_m * 16 + j + half * 8;
    out[(size_t)r * 256 + (tn0 + 0) * 16 + nloc] = acc0[j] + b_out[(tn0 + 0) * 16 + nloc];
    out[(size_t)r * 256 + (tn0 + 1) * 16 + nloc] = acc1[j] + b_out[(tn0 + 1) * 16 + nloc];
  }
}

// ---------------------------------------------------------------------------
extern "C" void kernel_launch(void* const* d_in, const int* in_sizes, int n_in,
                              void* d_out, int out_size, void* d_ws, size_t ws_size,
                              hipStream_t stream) {
  const float* x      = (const float*)d_in[0];  // [64,512,512]
  const float* noises = (const float*)d_in[1];  // [64,512,1024]
  const float* Wi     = (const float*)d_in[2];  // [4,1024,512]
  const float* bWi    = (const float*)d_in[3];  // [4,1024]
  const float* Wh     = (const float*)d_in[4];  // [4,1024,1024]
  const float* bWh    = (const float*)d_in[5];  // [4,1024]
  const float* bgate  = (const float*)d_in[6];  // [4,1024]
  const float* W_out  = (const float*)d_in[7];  // [256,1024]
  const float* b_out  = (const float*)d_in[8];  // [256]
  const float* b_h    = (const float*)d_in[9];  // [1,1024]

  float* out     = (float*)d_out;
  float* outputs = out;                                   // 64*512*256
  float* hidden  = out + (size_t)64 * 512 * 256;          // 64*512*1024
  float* l2w     = hidden + (size_t)64 * 512 * 1024;      // 4*1024*512

  // workspace layout (bytes)
  char* ws = (char*)d_ws;
  __bf16* Bp0  = (__bf16*)(ws + 0);                  // 2,097,152 el = 4 MB
  __bf16* Bp1  = (__bf16*)(ws + 4194304);            // 4,194,304 el = 8 MB
  __bf16* Bp2  = (__bf16*)(ws + 12582912);           //   262,144 el = 512 KB
  float*  bsum = (float*) (ws + 13107200);           // 4096 f32
  float*  hf0  = (float*) (ws + 13123584);           // 65536 f32
  float*  hf1  = (float*) (ws + 13385728);           // 65536 f32
  __bf16* hb0  = (__bf16*)(ws + 13647872);           // 65536 bf16
  __bf16* hb1  = (__bf16*)(ws + 13778944);           // 65536 bf16
  float*  cbuf = (float*) (ws + 13910016);           // 65536 f32

  // 1) pack weights into WMMA B-fragment order (bf16)
  pack_b_kernel<<<(2097152 + 255) / 256, 256, 0, stream>>>(Wi,    Bp0,  512, 256, 2097152);
  pack_b_kernel<<<(4194304 + 255) / 256, 256, 0, stream>>>(Wh,    Bp1, 1024, 256, 4194304);
  pack_b_kernel<<<( 262144 + 255) / 256, 256, 0, stream>>>(W_out, Bp2, 1024,  16,  262144);

  // 2) combine biases + zero recurrent state
  bias_init_kernel<<<256, 256, 0, stream>>>(bWi, bWh, bgate, bsum, hf0, hb0, cbuf);

  // 3) sequential scan: ONE fused kernel per step (GEMM + LSTM update),
  //    h double-buffered (ping-pong) across steps.
  float*  hfp[2] = {hf0, hf1};
  __bf16* hbp[2] = {hb0, hb1};
  for (int s = 0; s < 512; ++s) {
    step_fused_kernel<<<128, 256, 0, stream>>>(
        x, hbp[s & 1], Bp0, Bp1, bsum, noises, b_h,
        cbuf, hfp[s & 1], hfp[(s + 1) & 1], hbp[(s + 1) & 1], hidden, s);
  }

  // 4) output projection
  out_gemm_kernel<<<2048, 256, 0, stream>>>(hidden, Bp2, b_out, outputs);

  // 5) l2_weights = Wi flattened
  hipMemcpyAsync(l2w, Wi, sizeof(float) * 2097152, hipMemcpyDeviceToDevice, stream);
}
